// TransformerDecoderLinear_46016279610047
// MI455X (gfx1250) — matrix operand order repaired and checked by reading, not compiled, not run on hardware
//
#include <hip/hip_runtime.h>

typedef __bf16 bf16_t;
typedef __attribute__((ext_vector_type(16))) __bf16 v16bf;
typedef __attribute__((ext_vector_type(8)))  float  v8f;
typedef __attribute__((ext_vector_type(4)))  unsigned int v4u;
typedef __attribute__((ext_vector_type(8)))  int v8i;
typedef __attribute__((ext_vector_type(4)))  int v4i;

#if defined(__has_builtin)
#if __has_builtin(__builtin_amdgcn_tensor_load_to_lds) && __has_builtin(__builtin_amdgcn_s_wait_tensorcnt)
#define HAVE_TDM 1
#endif
#endif

// ---------------- problem constants ----------------
constexpr int Tq = 100;        // query time steps
constexpr int Sk = 4096;       // key/value time steps
constexpr int Bb = 32;         // batch
constexpr int Dd = 256;        // model dim
constexpr int NH = 8;          // heads
constexpr int HD = 32;         // head dim
constexpr int TTILES  = 7;     // ceil(100/16)
constexpr int STILES  = 256;   // 4096/16
constexpr int SCHUNKS = 128;   // 4096/32
constexpr int KCH     = 8;     // 256/32 K-chunks per GEMM
constexpr int NTILES  = 16;    // 256/16

// ---------------- workspace layout (bytes) ----------------
constexpr size_t SZ_WF = 4ull * NTILES * KCH * 512 * 2;            // 512 KB  (Wq,Wk,Wv,Wo bf16 B-frags)
constexpr size_t OFF_WF = 0;
constexpr size_t OFF_Q  = OFF_WF + SZ_WF;
constexpr size_t SZ_Q   = (size_t)Bb * NH * TTILES  * 512 * 2;     // 1.75 MB (Q A-frags)
constexpr size_t OFF_K  = OFF_Q + SZ_Q;
constexpr size_t SZ_K   = (size_t)Bb * NH * STILES  * 512 * 2;     // 64 MB  (K^T B-frags)
constexpr size_t OFF_V  = OFF_K + SZ_K;
constexpr size_t SZ_V   = (size_t)Bb * NH * SCHUNKS * 2 * 512 * 2; // 64 MB  (V B-frags)
constexpr size_t OFF_AO = OFF_V + SZ_V;                            // attention out f32 [T][B][D]

// ---- fragment index helpers (ISA 7.12.2, 16-bit operands, wave32) ----
// A-matrix 16x32: lane = hi*16 + m (hi selects K%16 >= 8), 16 bf16 per lane.
__device__ __forceinline__ int a_pos(int m, int k) {
  int hi = (k >> 3) & 1;
  int lane = hi * 16 + m;
  int j = (k >> 4) * 4 + ((k & 7) >> 1);
  return lane * 16 + j * 2 + (k & 1);
}
// B-matrix 32x16: lane = hi*16 + n (hi selects K%16 >= 8), same element order as A.
__device__ __forceinline__ int b_pos(int k, int n) {
  int lane = ((k >> 3) & 1) * 16 + n;
  int j = (k >> 4) * 4 + ((k & 7) >> 1);
  return lane * 16 + j * 2 + (k & 1);
}

// ---------------- kernel 0: pack weights into bf16 B-fragments ----------------
__global__ void pack_weights_kernel(const float* __restrict__ inW,
                                    const float* __restrict__ outW,
                                    char* __restrict__ wsb) {
  bf16_t* wf = (bf16_t*)(wsb + OFF_WF);
  int idx = blockIdx.x * blockDim.x + threadIdx.x;   // 4 * 256 * 256 = 262144 threads
  int mat = idx >> 16;
  int rem = idx & 65535;
  int n = rem >> 8;        // output dim (row of W)
  int k = rem & 255;       // input dim
  float v = (mat < 3) ? inW[(size_t)(mat * 256 + n) * 256 + k]
                      : outW[(size_t)n * 256 + k];
  int ntile = n >> 4, nn = n & 15, kc = k >> 5, kk = k & 31;
  wf[(size_t)mat * 65536 + (size_t)(ntile * KCH + kc) * 512 + b_pos(kk, nn)] = (bf16_t)v;
}

// ---------------- kernel 1: zero padded Q fragment region ----------------
__global__ void zero_qfrag_kernel(char* __restrict__ wsb) {
  unsigned* p = (unsigned*)(wsb + OFF_Q);
  size_t i = (size_t)blockIdx.x * blockDim.x + threadIdx.x;
  if (i < SZ_Q / 4) p[i] = 0u;
}

// ---------------- kernel 2: projection GEMM (X @ W^T + b) * scale ----------------
// mode 0: -> Q A-frags   mode 1: -> K^T B-frags   mode 2: -> V B-frags   mode 3: -> d_out f32
__global__ void __launch_bounds__(256)
proj_gemm_kernel(const float* __restrict__ X, char* __restrict__ wsb,
                 const float* __restrict__ bias, float scale,
                 int mode, int wmat, float* __restrict__ outO) {
  __shared__ float tx[16][256];   // 16 KB fp32 A tile
  const bf16_t* wf = (const bf16_t*)(wsb + OFF_WF) + (size_t)wmat * 65536;
  bf16_t* wq = (bf16_t*)(wsb + OFF_Q);
  bf16_t* wk = (bf16_t*)(wsb + OFF_K);
  bf16_t* wv = (bf16_t*)(wsb + OFF_V);

  int tid = threadIdx.x;
  int lane = tid & 31;
  int wave = tid >> 5;
  int rowTile = blockIdx.x;
  int ntile = blockIdx.y * 8 + wave;

#ifdef HAVE_TDM
  // ---- Tensor Data Mover: DMA the 16x256 fp32 tile into LDS (one issue per block) ----
  if (wave == 0) {
    unsigned long long gaddr =
        (unsigned long long)(const void*)(X + (size_t)rowTile * 16 * 256);
    // Flat shared addresses carry the LDS byte offset in bits [31:0] (aperture tag in [63:32]).
    unsigned int ldsoff = (unsigned int)(unsigned long long)(const void*)&tx[0][0];

    // D# group 0: count=1 | lds_addr | global_addr[56:0] | type=2 ("image")
    v4u g0;
    g0[0] = 1u;                                            // count = 1 valid descriptor
    g0[1] = ldsoff;                                        // lds_addr
    g0[2] = (unsigned int)(gaddr & 0xFFFFFFFFull);         // global_addr[31:0]
    g0[3] = (unsigned int)((gaddr >> 32) & 0x01FFFFFFull)  // global_addr[56:32]
            | (2u << 30);                                  // type = 2

    // D# group 1: data_size=4B, tensor/tile dims: dim0=256, dim1=16, dim0_stride=256
    v8i g1;
    g1[0] = (int)(2u << 16);          // workgroup_mask=0, data_size=2 (4 bytes)
    g1[1] = (int)(256u << 16);        // tensor_dim0[15:0] in bits [63:48]
    g1[2] = (int)(16u << 16);         // tensor_dim0[31:16]=0, tensor_dim1[15:0]=16
    g1[3] = (int)(256u << 16);        // tensor_dim1[31:16]=0, tile_dim0=256
    g1[4] = 16;                       // tile_dim1=16, tile_dim2=0 (unused)
    g1[5] = 256;                      // tensor_dim0_stride[31:0] = 256 elements
    g1[6] = 0;                        // dim0_stride hi, dim1_stride lo
    g1[7] = 0;                        // dim1_stride hi (unused: 2D tile)

    // D# groups 2/3: degenerate higher dims (dims=1, tiles unused)
    v4i g2; g2[0] = 1; g2[1] = 1; g2[2] = 0; g2[3] = 0;          // tensor_dim2=1, tensor_dim3=1
    v4i g3; g3[0] = 0; g3[1] = (int)(1u << 16); g3[2] = 0; g3[3] = 0; // tensor_dim4=1

#if __clang_major__ >= 23
    v8i gx = {};                      // clang-23 toolchain: 6-arg form
    __builtin_amdgcn_tensor_load_to_lds(g0, g1, g2, g3, gx, 0);
#else
    __builtin_amdgcn_tensor_load_to_lds(g0, g1, g2, g3, 0);       // ROCm 7.2 5-arg form
#endif
    __builtin_amdgcn_s_wait_tensorcnt(0);
  }
  __syncthreads();
#else
  // fallback: cooperative, coalesced load of the 16x256 fp32 A tile
  #pragma unroll
  for (int i = 0; i < 16; ++i) {
    int idx = i * 256 + tid;
    int r = idx >> 8, c = idx & 255;
    tx[r][c] = X[((size_t)rowTile * 16 + r) * 256 + c];
  }
  __syncthreads();
#endif

  int m = lane & 15;
  int hi = lane >> 4;
  v8f acc = {};
  #pragma unroll
  for (int kc = 0; kc < KCH; ++kc) {
    v16bf af;
    #pragma unroll
    for (int j = 0; j < 8; ++j)
      #pragma unroll
      for (int e = 0; e < 2; ++e) {
        int k = kc * 32 + (j >> 2) * 16 + hi * 8 + (j & 3) * 2 + e;
        af[j * 2 + e] = (bf16_t)tx[m][k];
      }
    v16bf bfg = *(const v16bf*)(wf + (size_t)(ntile * KCH + kc) * 512 + lane * 16);
    acc = __builtin_amdgcn_wmma_f32_16x16x32_bf16(false, af, false, bfg,
                                                  (short)0, acc, false, false);
  }

  // epilogue: bias + scale, scatter into consumer layout
  #pragma unroll
  for (int r = 0; r < 8; ++r) {
    int mm = r + 8 * hi;
    int grow = rowTile * 16 + mm;          // flat row: (t|s)*B + b
    int n = ntile * 16 + (lane & 15);
    float val = (acc[r] + bias[n]) * scale;
    if (mode == 3) {
      outO[(size_t)grow * 256 + n] = val;
    } else {
      int b = grow & 31;                   // B == 32
      int ts = grow >> 5;                  // t or s
      int h = n >> 5, hd = n & 31;
      if (mode == 0) {
        wq[(size_t)((b * NH + h) * TTILES + (ts >> 4)) * 512 + a_pos(ts & 15, hd)] = (bf16_t)val;
      } else if (mode == 1) {
        wk[(size_t)((b * NH + h) * STILES + (ts >> 4)) * 512 + b_pos(hd, ts & 15)] = (bf16_t)val;
      } else {
        wv[(size_t)(((b * NH + h) * SCHUNKS + (ts >> 5)) * 2 + (hd >> 4)) * 512
           + b_pos(ts & 31, hd & 15)] = (bf16_t)val;
      }
    }
  }
}

// ---------------- kernel 3: flash attention with post-softmax mask ----------------
__global__ void __launch_bounds__(256)
attention_kernel(char* __restrict__ wsb, const float* __restrict__ pmask) {
  __shared__ float plds[8][16][32];   // per-wave P transpose staging
  const bf16_t* wq = (const bf16_t*)(wsb + OFF_Q);
  const bf16_t* wk = (const bf16_t*)(wsb + OFF_K);
  const bf16_t* wv = (const bf16_t*)(wsb + OFF_V);
  float* wao = (float*)(wsb + OFF_AO);

  int lane = threadIdx.x & 31;
  int wave = threadIdx.x >> 5;
  int task = blockIdx.x * 8 + wave;         // B*NH*TTILES = 1792 tasks
  int b = task / (NH * TTILES);
  int rem = task % (NH * TTILES);
  int h = rem / TTILES;
  int ttile = rem % TTILES;
  int bh = b * NH + h;

  v16bf qf = *(const v16bf*)(wq + (size_t)(bh * TTILES + ttile) * 512 + lane * 16);

  int hi = lane >> 4;
  int nn = lane & 15;

  float rm[8], rz[8];
  #pragma unroll
  for (int r = 0; r < 8; ++r) { rm[r] = -1e30f; rz[r] = 0.f; }
  v8f acc0 = {}, acc1 = {};

  const bf16_t* kbh = wk + (size_t)bh * STILES * 512;
  const bf16_t* vbh = wv + (size_t)bh * SCHUNKS * 1024;

  for (int sc = 0; sc < SCHUNKS; ++sc) {
    const bf16_t* kbase = kbh + (size_t)sc * 1024;
    const bf16_t* vbase = vbh + (size_t)sc * 1024;
    if (sc + 1 < SCHUNKS) {
      __builtin_prefetch(kbase + 1024 + lane * 16, 0, 0);   // global_prefetch_b8
      __builtin_prefetch(vbase + 1024 + lane * 16, 0, 0);
    }
    v16bf k0 = *(const v16bf*)(kbase + lane * 16);
    v16bf k1 = *(const v16bf*)(kbase + 512 + lane * 16);
    v8f z0 = {}, z1 = {};
    v8f s0 = __builtin_amdgcn_wmma_f32_16x16x32_bf16(false, qf, false, k0, (short)0, z0, false, false);
    v8f s1 = __builtin_amdgcn_wmma_f32_16x16x32_bf16(false, qf, false, k1, (short)0, z1, false, false);

    // row max over the 32 S-columns: butterfly within each 16-lane half
    float tmax[8];
    #pragma unroll
    for (int r = 0; r < 8; ++r) tmax[r] = fmaxf(s0[r], s1[r]);
    #pragma unroll
    for (int msk = 1; msk < 16; msk <<= 1)
      #pragma unroll
      for (int r = 0; r < 8; ++r)
        tmax[r] = fmaxf(tmax[r], __shfl_xor(tmax[r], msk, 32));

    float e0[8], e1[8], alpha[8], tsum[8];
    #pragma unroll
    for (int r = 0; r < 8; ++r) {
      float nm = fmaxf(rm[r], tmax[r]);
      alpha[r] = __expf(rm[r] - nm);
      rm[r] = nm;
      e0[r] = __expf(s0[r] - nm);
      e1[r] = __expf(s1[r] - nm);
      tsum[r] = e0[r] + e1[r];
    }
    #pragma unroll
    for (int msk = 1; msk < 16; msk <<= 1)
      #pragma unroll
      for (int r = 0; r < 8; ++r)
        tsum[r] += __shfl_xor(tsum[r], msk, 32);
    #pragma unroll
    for (int r = 0; r < 8; ++r) rz[r] = rz[r] * alpha[r] + tsum[r];   // Z excludes mask

    // numerator includes mask (reference: attn = softmax(scores) * mask)
    #pragma unroll
    for (int r = 0; r < 8; ++r) {
      int mm = r + 8 * hi;
      int t = ttile * 16 + mm;
      float m0 = 0.f, m1 = 0.f;
      if (t < Tq) {
        const float* mp = pmask + ((size_t)b * Tq + t) * Sk + (size_t)sc * 32;
        m0 = mp[nn];
        m1 = mp[16 + nn];
      }
      plds[wave][mm][nn]      = e0[r] * m0;
      plds[wave][mm][16 + nn] = e1[r] * m1;
    }
    asm volatile("s_wait_dscnt 0" ::: "memory");   // wave-local LDS RAW fence

    v16bf pf;
    #pragma unroll
    for (int j = 0; j < 8; ++j)
      #pragma unroll
      for (int e = 0; e < 2; ++e) {
        int k = (j >> 2) * 16 + hi * 8 + (j & 3) * 2 + e;
        pf[j * 2 + e] = (bf16_t)plds[wave][lane & 15][k];
      }

    #pragma unroll
    for (int r = 0; r < 8; ++r) { acc0[r] *= alpha[r]; acc1[r] *= alpha[r]; }

    v16bf v0 = *(const v16bf*)(vbase + lane * 16);
    v16bf v1 = *(const v16bf*)(vbase + 512 + lane * 16);
    acc0 = __builtin_amdgcn_wmma_f32_16x16x32_bf16(false, pf, false, v0, (short)0, acc0, false, false);
    acc1 = __builtin_amdgcn_wmma_f32_16x16x32_bf16(false, pf, false, v1, (short)0, acc1, false, false);
  }

  #pragma unroll
  for (int r = 0; r < 8; ++r) {
    int mm = r + 8 * hi;
    int t = ttile * 16 + mm;
    if (t < Tq) {
      float inv = 1.f / rz[r];
      size_t rowoff = ((size_t)t * Bb + b) * Dd + h * HD;
      wao[rowoff + nn]      = acc0[r] * inv;
      wao[rowoff + 16 + nn] = acc1[r] * inv;
    }
  }
}

// ---------------- host launch ----------------
extern "C" void kernel_launch(void* const* d_in, const int* in_sizes, int n_in,
                              void* d_out, int out_size, void* d_ws, size_t ws_size,
                              hipStream_t stream) {
  const float* query = (const float*)d_in[0];
  const float* key   = (const float*)d_in[1];
  const float* value = (const float*)d_in[2];
  const float* pmask = (const float*)d_in[3];
  const float* inW   = (const float*)d_in[4];
  const float* inB   = (const float*)d_in[5];
  const float* outW  = (const float*)d_in[6];
  const float* outB  = (const float*)d_in[7];
  float* out = (float*)d_out;
  char* ws = (char*)d_ws;

  pack_weights_kernel<<<1024, 256, 0, stream>>>(inW, outW, ws);
  zero_qfrag_kernel<<<(int)((SZ_Q / 4 + 255) / 256), 256, 0, stream>>>(ws);

  const float qscale = 0.17677669529663689f;   // hd^-0.5, hd = 32
  proj_gemm_kernel<<<dim3( 200, 2), 256, 0, stream>>>(query, ws, inB,        qscale, 0, 0, nullptr);
  proj_gemm_kernel<<<dim3(8192, 2), 256, 0, stream>>>(key,   ws, inB + 256,  1.0f,   1, 1, nullptr);
  proj_gemm_kernel<<<dim3(8192, 2), 256, 0, stream>>>(value, ws, inB + 512,  1.0f,   2, 2, nullptr);

  attention_kernel<<<224, 256, 0, stream>>>(ws, pmask);

  proj_gemm_kernel<<<dim3(200, 2), 256, 0, stream>>>((const float*)(ws + OFF_AO),
                                                     ws, outB, 1.0f, 3, 3, out);
}
